// DecoderLayer_66039417143308
// MI455X (gfx1250) — matrix operand order repaired and checked by reading, not compiled, not run on hardware
//
#include <hip/hip_runtime.h>
#include <hip/hip_bf16.h>

// ---------------------------------------------------------------------------
// CDNA5 async global->LDS support (probe via __has_builtin; safe fallback)
// ---------------------------------------------------------------------------
#if defined(__has_builtin)
# if __has_builtin(__builtin_amdgcn_global_load_async_to_lds_b128)
#  define HAVE_ASYNC 1
# endif
#endif
#ifndef HAVE_ASYNC
# define HAVE_ASYNC 0
#endif

#if HAVE_ASYNC
typedef int async_v4i __attribute__((vector_size(16)));
// 16-byte-per-lane async DMA: LDS[l] = MEM[g]; tracked by ASYNCcnt.
__device__ __forceinline__ void async_cp16(const unsigned short* g, unsigned short* l) {
    __builtin_amdgcn_global_load_async_to_lds_b128(
        (__attribute__((address_space(1))) async_v4i*)(void*)g,
        (__attribute__((address_space(3))) async_v4i*)(void*)l,
        0, 0);
}
__device__ __forceinline__ void wait_async0() {
    asm volatile("s_wait_asynccnt 0" ::: "memory");
}
#endif

// ---------------------------------------------------------------------------
// Types / helpers
// ---------------------------------------------------------------------------
typedef __attribute__((ext_vector_type(16))) __bf16 v16bf;
typedef __attribute__((ext_vector_type(8)))  float  v8f;

__device__ __forceinline__ unsigned short f2bfu(float f) {
    unsigned u = __float_as_uint(f);
    u += 0x7FFFu + ((u >> 16) & 1u);           // round-to-nearest-even
    return (unsigned short)(u >> 16);
}
__device__ __forceinline__ v8f v8zero() {
    v8f z;
#pragma unroll
    for (int i = 0; i < 8; ++i) z[i] = 0.0f;
    return z;
}

union VB16 { unsigned w[8]; v16bf v; };

// Build a 16-element bf16 fragment from two contiguous 16-byte chunks.
__device__ __forceinline__ v16bf ldfrag(const unsigned short* p0,
                                        const unsigned short* p1) {
    uint4 a = *(const uint4*)p0;
    uint4 b = *(const uint4*)p1;
    VB16 t;
    t.w[0] = a.x; t.w[1] = a.y; t.w[2] = a.z; t.w[3] = a.w;
    t.w[4] = b.x; t.w[5] = b.y; t.w[6] = b.z; t.w[7] = b.w;
    return t.v;
}

// ---------------------------------------------------------------------------
// Elementwise f32 -> bf16 (vector x4)
// ---------------------------------------------------------------------------
__global__ __launch_bounds__(256)
void cvt_bf16_kernel(const float* __restrict__ in, unsigned short* __restrict__ out,
                     long n4)
{
    long i = (long)blockIdx.x * 256 + threadIdx.x;
    if (i >= n4) return;
    float4 v = ((const float4*)in)[i];
    unsigned u0 = (unsigned)f2bfu(v.x) | ((unsigned)f2bfu(v.y) << 16);
    unsigned u1 = (unsigned)f2bfu(v.z) | ((unsigned)f2bfu(v.w) << 16);
    ((uint2*)out)[i] = make_uint2(u0, u1);
}

// ---------------------------------------------------------------------------
// Weight transpose + convert: W[K,N] f32 -> Wt[N,K] bf16 (32x32 LDS tiles)
// ---------------------------------------------------------------------------
__global__ __launch_bounds__(256)
void transpose_cvt_kernel(const float* __restrict__ W, unsigned short* __restrict__ Wt,
                          int K, int N)
{
    __shared__ float t[32][33];
    const int tx = threadIdx.x & 31, ty = threadIdx.x >> 5;   // ty 0..7
    const long k0 = (long)blockIdx.y * 32, n0 = (long)blockIdx.x * 32;
#pragma unroll
    for (int j = 0; j < 4; ++j)
        t[ty + 8 * j][tx] = W[(k0 + ty + 8 * j) * N + n0 + tx];
    __syncthreads();
#pragma unroll
    for (int j = 0; j < 4; ++j)
        Wt[(n0 + ty + 8 * j) * (long)K + k0 + tx] = f2bfu(t[tx][ty + 8 * j]);
}

// ---------------------------------------------------------------------------
// bf16-WMMA GEMM: C[M,N] = act(A[M,K] @ B + bias),  B given as Bt[N,K] bf16.
// 256 threads = 8 waves; block tile 128x128x32; wave tile 32x64 (2x4 frags).
// Async-DMA double-buffered K pipeline when the toolchain exposes it.
// ---------------------------------------------------------------------------
#define GST 40   // LDS row stride in ushort (80 B, 16B aligned)

__global__ __launch_bounds__(256)
void gemm_bf16(const unsigned short* __restrict__ A,
               const unsigned short* __restrict__ Bt,
               const float* __restrict__ bias,
               float* __restrict__ Cf, unsigned short* __restrict__ Cb,
               int M, int N, int K, int relu)
{
    __shared__ unsigned short As[2][128 * GST];   // [m][k]
    __shared__ unsigned short Bs[2][128 * GST];   // [n][k]

    const int tid  = threadIdx.x;
    const int lane = tid & 31;
    const int wv   = tid >> 5;
    const int hi   = lane >> 4;
    const int l15  = lane & 15;
    const int wm   = wv >> 1;            // 0..3
    const int wn   = wv & 1;             // 0..1

    const long row0 = (long)blockIdx.y * 128;
    const long col0 = (long)blockIdx.x * 128;

    v8f acc[2][4];
#pragma unroll
    for (int mi = 0; mi < 2; ++mi)
#pragma unroll
        for (int ni = 0; ni < 4; ++ni) acc[mi][ni] = v8zero();

#if HAVE_ASYNC
    // async DMA of one 128x32 bf16 tile pair into buffer `buf`
    auto issue_tile = [&](int buf, int k0) {
#pragma unroll
        for (int s = 0; s < 2; ++s) {
            int i  = tid + s * 256;          // 0..511
            int r  = i >> 2;
            int c8 = (i & 3) * 8;
            async_cp16(A  + (row0 + r) * (long)K + k0 + c8, &As[buf][r * GST + c8]);
            async_cp16(Bt + (col0 + r) * (long)K + k0 + c8, &Bs[buf][r * GST + c8]);
        }
    };
    const int nt = K / 32;
    issue_tile(0, 0);
    for (int t = 0; t < nt; ++t) {
        wait_async0();          // tile t resident (own wave's DMAs done)
        __syncthreads();        // ...and everyone else's
        if (t + 1 < nt) issue_tile((t + 1) & 1, (t + 1) * 32);   // overlap w/ WMMA
        const int b = t & 1;
#else
    for (int k0 = 0; k0 < K; k0 += 32) {
        const int b = 0;
#pragma unroll
        for (int s = 0; s < 2; ++s) {
            int i  = tid + s * 256;
            int r  = i >> 2;
            int c8 = (i & 3) * 8;
            uint4 av = *(const uint4*)(A  + (row0 + r) * (long)K + k0 + c8);
            uint4 bv = *(const uint4*)(Bt + (col0 + r) * (long)K + k0 + c8);
            *(uint4*)(As[0] + r * GST + c8) = av;
            *(uint4*)(Bs[0] + r * GST + c8) = bv;
        }
        if (k0 + 32 < K) {
            __builtin_prefetch(A  + (row0 + (tid >> 1)) * (long)K + k0 + 32, 0, 3);
            __builtin_prefetch(Bt + (col0 + (tid >> 1)) * (long)K + k0 + 32, 0, 3);
        }
        __syncthreads();
#endif
        v16bf af[2];
#pragma unroll
        for (int mi = 0; mi < 2; ++mi) {
            const unsigned short* rp = As[b] + (wm * 32 + mi * 16 + l15) * GST;
            af[mi] = ldfrag(rp + 8 * hi, rp + 16 + 8 * hi);
        }
        v16bf bfr[4];
#pragma unroll
        for (int ni = 0; ni < 4; ++ni) {
            const unsigned short* rp = Bs[b] + (wn * 64 + ni * 16 + l15) * GST;
            bfr[ni] = ldfrag(rp + 8 * hi, rp + 16 + 8 * hi);
        }
#pragma unroll
        for (int mi = 0; mi < 2; ++mi)
#pragma unroll
            for (int ni = 0; ni < 4; ++ni)
                acc[mi][ni] = __builtin_amdgcn_wmma_f32_16x16x32_bf16(
                    false, af[mi], false, bfr[ni], (short)0, acc[mi][ni],
                    false, false);
#if !HAVE_ASYNC
        __syncthreads();
#endif
    }

#pragma unroll
    for (int mi = 0; mi < 2; ++mi) {
#pragma unroll
        for (int ni = 0; ni < 4; ++ni) {
            const long cc = col0 + wn * 64 + ni * 16 + l15;
            const float bv = bias[cc];
#pragma unroll
            for (int r = 0; r < 8; ++r) {
                const long rr = row0 + wm * 32 + mi * 16 + r + hi * 8;
                float v = acc[mi][ni][r] + bv;
                if (relu) v = fmaxf(v, 0.0f);
                const long idx = rr * (long)N + cc;
                if (Cf) Cf[idx] = v;
                if (Cb) Cb[idx] = f2bfu(v);
            }
        }
    }
}

// ---------------------------------------------------------------------------
// Flash attention (online softmax), all-bf16 I/O, hd = 64.
// Grid: (B*H, S_tgt/64); block = 128 = 4 waves; wave owns 16 q rows.
// ---------------------------------------------------------------------------
#define KVST 72   // Ks row stride (ushort): [key][d]
#define VTST 40   // Vt row stride (ushort): [d][key]
#define PST  40   // P  row stride (ushort): [row][key]

template <bool CAUSAL>
__global__ __launch_bounds__(128)
void flash_attn(const unsigned short* __restrict__ Q,
                const unsigned short* __restrict__ Kg,
                const unsigned short* __restrict__ Vg,
                unsigned short* __restrict__ O,
                int H, int S, int Ssrc)
{
    constexpr int HD = 64;
    const int D = H * HD;

    __shared__ unsigned short Ks[32 * KVST];
    __shared__ unsigned short Vt[64 * VTST];
    __shared__ unsigned short Ps[4][16 * PST];

    const int tid  = threadIdx.x;
    const int lane = tid & 31;
    const int wv   = tid >> 5;
    const int hi   = lane >> 4;
    const int l15  = lane & 15;

    const int bh   = blockIdx.x;
    const int b    = bh / H;
    const int h    = bh % H;
    const int qblk = blockIdx.y;
    const int q0   = qblk * 64 + wv * 16;

    // Q fragments (16x64 -> two K=32 A-frags)
    v16bf qa[2];
    {
        const unsigned short* qp = Q + ((long)b * S + q0 + l15) * D + h * HD;
#pragma unroll
        for (int f = 0; f < 2; ++f)
            qa[f] = ldfrag(qp + f * 32 + 8 * hi, qp + f * 32 + 16 + 8 * hi);
    }

    float m8[8], l8[8];
    v8f o[4];
#pragma unroll
    for (int r = 0; r < 8; ++r) { m8[r] = -3.0e38f; l8[r] = 0.0f; }
#pragma unroll
    for (int nf = 0; nf < 4; ++nf) o[nf] = v8zero();

    const float sc = 0.125f;   // 1/sqrt(64)
    const int kend = CAUSAL ? (qblk * 64 + 64) : Ssrc;   // uniform per block

    for (int kb = 0; kb < kend; kb += 32) {
        __syncthreads();
#pragma unroll
        for (int s = 0; s < 2; ++s) {
            int i  = tid + s * 128;          // 0..255 (32 keys x 8 chunks)
            int r  = i >> 3;                 // key row
            int c8 = (i & 7) * 8;            // d chunk
            long g = ((long)b * Ssrc + kb + r) * D + h * HD + c8;
#if HAVE_ASYNC
            async_cp16(Kg + g, &Ks[r * KVST + c8]);        // [key][d] via DMA
#else
            *(uint4*)(Ks + r * KVST + c8) = *(const uint4*)(Kg + g);
#endif
            uint4 vvec = *(const uint4*)(Vg + g);
            union { uint4 q; unsigned short s8[8]; } u; u.q = vvec;
#pragma unroll
            for (int j = 0; j < 8; ++j)                     // [d][key] transpose
                Vt[(c8 + j) * VTST + r] = u.s8[j];
        }
#if HAVE_ASYNC
        wait_async0();
#endif
        __syncthreads();

        // S = Q @ K^T  (two 16-key subtiles)
        v8f sf[2];
#pragma unroll
        for (int sub = 0; sub < 2; ++sub) {
            sf[sub] = v8zero();
            const unsigned short* kp = Ks + (sub * 16 + l15) * KVST;
            v16bf bk0 = ldfrag(kp + 8 * hi,      kp + 16 + 8 * hi);
            v16bf bk1 = ldfrag(kp + 32 + 8 * hi, kp + 48 + 8 * hi);
            sf[sub] = __builtin_amdgcn_wmma_f32_16x16x32_bf16(
                false, qa[0], false, bk0, (short)0, sf[sub], false, false);
            sf[sub] = __builtin_amdgcn_wmma_f32_16x16x32_bf16(
                false, qa[1], false, bk1, (short)0, sf[sub], false, false);
        }

        // mask + online softmax stats
        float mnew[8];
#pragma unroll
        for (int r = 0; r < 8; ++r) {
            float s0 = sf[0][r] * sc, s1 = sf[1][r] * sc;
            if (CAUSAL) {
                const int qg = q0 + r + hi * 8;
                if (kb + l15 > qg)      s0 = -1.0e30f;
                if (kb + 16 + l15 > qg) s1 = -1.0e30f;
            }
            sf[0][r] = s0; sf[1][r] = s1;
            float mx = fmaxf(s0, s1);
#pragma unroll
            for (int off = 1; off < 16; off <<= 1)
                mx = fmaxf(mx, __shfl_xor(mx, off, 32));
            mnew[r] = fmaxf(m8[r], mx);
        }
        float alpha[8];
#pragma unroll
        for (int r = 0; r < 8; ++r) {
            alpha[r] = __expf(m8[r] - mnew[r]);
            m8[r] = mnew[r];
            l8[r] *= alpha[r];
        }
#pragma unroll
        for (int nf = 0; nf < 4; ++nf)
#pragma unroll
            for (int r = 0; r < 8; ++r) o[nf][r] *= alpha[r];

        // P = exp(S - m) -> LDS (C->A layout hop); row sums
#pragma unroll
        for (int r = 0; r < 8; ++r) {
            float p0 = __expf(sf[0][r] - mnew[r]);
            float p1 = __expf(sf[1][r] - mnew[r]);
            const int prow = (r + hi * 8) * PST;
            Ps[wv][prow + l15]      = f2bfu(p0);
            Ps[wv][prow + 16 + l15] = f2bfu(p1);
            float ps = p0 + p1;
#pragma unroll
            for (int off = 1; off < 16; off <<= 1)
                ps += __shfl_xor(ps, off, 32);
            l8[r] += ps;
        }
        __syncthreads();   // uniform trip count; publish P for cross-lane reads

        // O += P @ V
        const unsigned short* pp = Ps[wv] + l15 * PST;
        v16bf pa = ldfrag(pp + 8 * hi, pp + 16 + 8 * hi);
#pragma unroll
        for (int nf = 0; nf < 4; ++nf) {
            const unsigned short* vp = Vt + (nf * 16 + l15) * VTST;
            v16bf bv = ldfrag(vp + 8 * hi, vp + 16 + 8 * hi);
            o[nf] = __builtin_amdgcn_wmma_f32_16x16x32_bf16(
                false, pa, false, bv, (short)0, o[nf], false, false);
        }
    }

    // epilogue: O /= l -> bf16
#pragma unroll
    for (int r = 0; r < 8; ++r) {
        const float inv = 1.0f / l8[r];
        const long rbase = ((long)b * S + q0 + r + hi * 8) * D + h * HD;
#pragma unroll
        for (int nf = 0; nf < 4; ++nf)
            O[rbase + nf * 16 + l15] = f2bfu(o[nf][r] * inv);
    }
}

// ---------------------------------------------------------------------------
// Residual + LayerNorm; optional bf16 mirror output for downstream GEMMs.
// ---------------------------------------------------------------------------
__global__ __launch_bounds__(256)
void ln_residual(const float* __restrict__ x, const float* __restrict__ res,
                 const float* __restrict__ g, const float* __restrict__ bb,
                 float* __restrict__ out, unsigned short* __restrict__ obf,
                 int D)
{
    __shared__ float red[8];
    __shared__ float stat;

    const long row = blockIdx.x;
    const int tid  = threadIdx.x;
    const int lane = tid & 31, wv = tid >> 5;
    const float* xr = x   + row * (long)D;
    const float* rr = res + row * (long)D;

    float v[4];
    float s = 0.0f;
#pragma unroll
    for (int i = 0; i < 4; ++i) {
        int c = tid + i * 256;
        v[i] = xr[c] + rr[c];
        s += v[i];
    }
#pragma unroll
    for (int off = 16; off >= 1; off >>= 1) s += __shfl_xor(s, off, 32);
    if (lane == 0) red[wv] = s;
    __syncthreads();
    if (tid == 0) {
        float t = 0.0f;
        for (int j = 0; j < 8; ++j) t += red[j];
        stat = t / (float)D;
    }
    __syncthreads();
    const float mu = stat;

    float s2 = 0.0f;
#pragma unroll
    for (int i = 0; i < 4; ++i) { float d = v[i] - mu; s2 += d * d; }
#pragma unroll
    for (int off = 16; off >= 1; off >>= 1) s2 += __shfl_xor(s2, off, 32);
    __syncthreads();
    if (lane == 0) red[wv] = s2;
    __syncthreads();
    if (tid == 0) {
        float t = 0.0f;
        for (int j = 0; j < 8; ++j) t += red[j];
        stat = t / (float)D;
    }
    __syncthreads();
    const float rstd = rsqrtf(stat + 1e-5f);

    float* orow = out + row * (long)D;
#pragma unroll
    for (int i = 0; i < 4; ++i) {
        int c = tid + i * 256;
        float y = (v[i] - mu) * rstd * g[c] + bb[c];
        orow[c] = y;
        if (obf) obf[row * (long)D + c] = f2bfu(y);
    }
}

// ---------------------------------------------------------------------------
// Orchestration
// ---------------------------------------------------------------------------
extern "C" void kernel_launch(void* const* d_in, const int* in_sizes, int n_in,
                              void* d_out, int out_size, void* d_ws, size_t ws_size,
                              hipStream_t stream)
{
    constexpr int B = 8, S = 1024, D = 1024, H = 16, DFF = 4096;
    constexpr long M = (long)B * S;                        // 8192 rows
    constexpr size_t F32 = (size_t)B * S * D * 4;          // 32 MB
    constexpr size_t BF  = (size_t)B * S * D * 2;          // 16 MB
    constexpr size_t WBF = (size_t)D * D * 2;              //  2 MB
    constexpr size_t WFF = (size_t)D * DFF * 2;            //  8 MB

    const float* x   = (const float*)d_in[0];
    const float* enc = (const float*)d_in[1];
    const float* sa_Wq = (const float*)d_in[4];  const float* sa_bq = (const float*)d_in[5];
    const float* sa_Wk = (const float*)d_in[6];  const float* sa_bk = (const float*)d_in[7];
    const float* sa_Wv = (const float*)d_in[8];  const float* sa_bv = (const float*)d_in[9];
    const float* sa_Wo = (const float*)d_in[10]; const float* sa_bo = (const float*)d_in[11];
    const float* ca_Wq = (const float*)d_in[12]; const float* ca_bq = (const float*)d_in[13];
    const float* ca_Wk = (const float*)d_in[14]; const float* ca_bk = (const float*)d_in[15];
    const float* ca_Wv = (const float*)d_in[16]; const float* ca_bv = (const float*)d_in[17];
    const float* ca_Wo = (const float*)d_in[18]; const float* ca_bo = (const float*)d_in[19];
    const float* ff_W1 = (const float*)d_in[20]; const float* ff_b1 = (const float*)d_in[21];
    const float* ff_W2 = (const float*)d_in[22]; const float* ff_b2 = (const float*)d_in[23];
    const float* ln1_g = (const float*)d_in[24]; const float* ln1_b = (const float*)d_in[25];
    const float* ln2_g = (const float*)d_in[26]; const float* ln2_b = (const float*)d_in[27];
    const float* ln3_g = (const float*)d_in[28]; const float* ln3_b = (const float*)d_in[29];
    float* out = (float*)d_out;

    // ---- workspace carve-up ----
    char* p = (char*)d_ws;
    unsigned short* xbf   = (unsigned short*)p; p += BF;
    unsigned short* encbf = (unsigned short*)p; p += BF;
    unsigned short* wT[8];
    for (int i = 0; i < 8; ++i) { wT[i] = (unsigned short*)p; p += WBF; }  // sa q,k,v,o | ca q,k,v,o
    unsigned short* w1T = (unsigned short*)p; p += WFF;   // [DFF, D]
    unsigned short* w2T = (unsigned short*)p; p += WFF;   // [D, DFF]
    unsigned short* Qb    = (unsigned short*)p; p += BF;
    unsigned short* Kb    = (unsigned short*)p; p += BF;
    unsigned short* Vb    = (unsigned short*)p; p += BF;
    unsigned short* attnb = (unsigned short*)p; p += BF;
    unsigned short* x1b   = (unsigned short*)p; p += BF;
    unsigned short* x2b   = (unsigned short*)p; p += BF;
    unsigned short* hidb  = (unsigned short*)p; p += (size_t)B * S * DFF * 2;  // 64 MB
    float* proj = (float*)p; p += F32;
    float* x1f  = (float*)p; p += F32;
    float* x2f  = (float*)p; p += F32;
    float* ffo  = (float*)p; p += F32;

    const dim3 blkG(256), blkF(128), blkL(256), blkC(256), blkT(256);
    const dim3 grdProj(D / 128, M / 128);       // 8  x 64
    const dim3 grdFF1(DFF / 128, M / 128);      // 32 x 64
    const dim3 grdFlash(B * H, S / 64);         // 128 x 16
    const dim3 grdLN((unsigned)M);
    const dim3 grdCvt((unsigned)((M * D) / 4 / 256));
    const dim3 grdTsq(D / 32, D / 32);
    const dim3 grdT1(DFF / 32, D / 32);         // W1: K=D rows, N=DFF cols
    const dim3 grdT2(D / 32, DFF / 32);         // W2: K=DFF rows, N=D cols

    // ---- one-time conversions (per launch) ----
    cvt_bf16_kernel<<<grdCvt, blkC, 0, stream>>>(x,   xbf,   (M * D) / 4);
    cvt_bf16_kernel<<<grdCvt, blkC, 0, stream>>>(enc, encbf, (M * D) / 4);
    const float* wsrc[8] = {sa_Wq, sa_Wk, sa_Wv, sa_Wo, ca_Wq, ca_Wk, ca_Wv, ca_Wo};
    for (int i = 0; i < 8; ++i)
        transpose_cvt_kernel<<<grdTsq, blkT, 0, stream>>>(wsrc[i], wT[i], D, D);
    transpose_cvt_kernel<<<grdT1, blkT, 0, stream>>>(ff_W1, w1T, D, DFF);
    transpose_cvt_kernel<<<grdT2, blkT, 0, stream>>>(ff_W2, w2T, DFF, D);

    // ---- self-attention ----
    gemm_bf16<<<grdProj, blkG, 0, stream>>>(xbf, wT[0], sa_bq, nullptr, Qb, M, D, D, 0);
    gemm_bf16<<<grdProj, blkG, 0, stream>>>(xbf, wT[1], sa_bk, nullptr, Kb, M, D, D, 0);
    gemm_bf16<<<grdProj, blkG, 0, stream>>>(xbf, wT[2], sa_bv, nullptr, Vb, M, D, D, 0);
    flash_attn<true><<<grdFlash, blkF, 0, stream>>>(Qb, Kb, Vb, attnb, H, S, S);
    gemm_bf16<<<grdProj, blkG, 0, stream>>>(attnb, wT[3], sa_bo, proj, nullptr, M, D, D, 0);
    ln_residual<<<grdLN, blkL, 0, stream>>>(x, proj, ln1_g, ln1_b, x1f, x1b, D);

    // ---- cross-attention ----
    gemm_bf16<<<grdProj, blkG, 0, stream>>>(x1b,   wT[4], ca_bq, nullptr, Qb, M, D, D, 0);
    gemm_bf16<<<grdProj, blkG, 0, stream>>>(encbf, wT[5], ca_bk, nullptr, Kb, M, D, D, 0);
    gemm_bf16<<<grdProj, blkG, 0, stream>>>(encbf, wT[6], ca_bv, nullptr, Vb, M, D, D, 0);
    flash_attn<false><<<grdFlash, blkF, 0, stream>>>(Qb, Kb, Vb, attnb, H, S, S);
    gemm_bf16<<<grdProj, blkG, 0, stream>>>(attnb, wT[7], ca_bo, proj, nullptr, M, D, D, 0);
    ln_residual<<<grdLN, blkL, 0, stream>>>(x1f, proj, ln2_g, ln2_b, x2f, x2b, D);

    // ---- FFN ----
    gemm_bf16<<<grdFF1, blkG, 0, stream>>>(x2b, w1T, ff_b1, nullptr, hidb, M, DFF, D, 1);
    gemm_bf16<<<grdProj, blkG, 0, stream>>>(hidb, w2T, ff_b2, ffo, nullptr, M, D, DFF, 0);
    ln_residual<<<grdLN, blkL, 0, stream>>>(x2f, ffo, ln3_g, ln3_b, out, nullptr, D);
}
